// GAT_20461224198294
// MI455X (gfx1250) — matrix operand order repaired and checked by reading, not compiled, not run on hardware
//
#include <hip/hip_runtime.h>
#include <hip/hip_bf16.h>

// ---------------------------------------------------------------------------
// GAT (dense all-pairs attention), MI455X / gfx1250, wave32 + WMMA bf16
// + async global->LDS double buffering (ASYNCcnt) in both GEMMs
// + sched_barrier-enforced load/WMMA batching for staggered waits.
//   x:[4096,512] f32, W:[4,512,128] f32, a:[4,256] f32 -> out:[4096,512] f32
// ---------------------------------------------------------------------------

#define NN      4096   // nodes
#define FIN     512    // input features
#define FH      128    // hidden per head
#define NH      4      // heads
#define LRELU   0.2f

typedef __attribute__((ext_vector_type(16))) __bf16 v16bf;
typedef __attribute__((ext_vector_type(8)))  __bf16 v8bf;
typedef __attribute__((ext_vector_type(8)))  float  v8f;

__device__ __forceinline__ v16bf cat8(v8bf lo, v8bf hi) {
    v16bf r;
#pragma unroll
    for (int i = 0; i < 8; ++i) { r[i] = lo[i]; r[8 + i] = hi[i]; }
    return r;
}

__device__ __forceinline__ void wait_async0() {
#if __has_builtin(__builtin_amdgcn_s_wait_asynccnt)
    __builtin_amdgcn_s_wait_asynccnt(0);
#else
    asm volatile("s_wait_asynccnt 0" ::: "memory");
#endif
}

__device__ __forceinline__ void sched_fence() {
#if __has_builtin(__builtin_amdgcn_sched_barrier)
    __builtin_amdgcn_sched_barrier(0);   // keep loads clustered before WMMAs
#endif
}

// One wave32 instruction copies 32 lanes x 16B = 512B global -> LDS (ASYNCcnt).
__device__ __forceinline__ void async_g2l_b128(const void* gaddr, void* lds) {
    unsigned           l = (unsigned)(unsigned long long)(uintptr_t)lds; // low 32b = LDS offset
    unsigned long long g = (unsigned long long)(uintptr_t)gaddr;
    asm volatile("global_load_async_to_lds_b128 %0, %1, off" :: "v"(l), "v"(g) : "memory");
}

// --------------------------- prep kernels ----------------------------------
__global__ void prep_x_kernel(const float* __restrict__ x, __bf16* __restrict__ xb) {
    for (int i = blockIdx.x * blockDim.x + threadIdx.x; i < NN * FIN;
         i += gridDim.x * blockDim.x)
        xb[i] = (__bf16)x[i];
}

// Wt[h][c][k] = bf16(W[h][k][c])  (K-contiguous so B fragments are contiguous)
__global__ void prep_w_kernel(const float* __restrict__ W, __bf16* __restrict__ wt) {
    for (int i = blockIdx.x * blockDim.x + threadIdx.x; i < NH * FH * FIN;
         i += gridDim.x * blockDim.x) {
        int k = i % FIN;
        int c = (i / FIN) % FH;
        int h = i / (FIN * FH);
        wt[i] = (__bf16)W[((size_t)h * FIN + k) * FH + c];
    }
}

// --------------------- kernel A: h = x @ W (+ f_src/f_dst) -----------------
// 128 threads = 4 waves; wave handles 16 rows x 128 cols; grid (NN/64, NH).
// Wt tile (128x32 bf16 = 8KB) double-buffered in LDS via async loads; A
// fragment software-pipelined one K-step ahead in registers.
__global__ __launch_bounds__(128)
void gat_h_kernel(const __bf16* __restrict__ xb, const __bf16* __restrict__ wt,
                  const float* __restrict__ a, __bf16* __restrict__ hbT,
                  float* __restrict__ fsrc, float* __restrict__ fdst) {
    __shared__ __align__(64) __bf16 s_w[2][FH * 32];   // 2 x 8 KB ping-pong

    const int tid  = threadIdx.x;
    const int lane = tid & 31;
    const int wave = tid >> 5;
    const int head = blockIdx.y;
    const int row0 = blockIdx.x * 64 + wave * 16;
    const int half = lane >> 4;   // 0: lanes 0-15, 1: lanes 16-31
    const int lm   = lane & 15;

    const __bf16* xrow = xb + (size_t)(row0 + lm) * FIN + half * 16;  // A frag base
    const __bf16* wth  = wt + (size_t)head * FH * FIN;

    const int c0 = tid >> 2, ch = (tid & 3) * 8;       // copy slice per thread
#pragma unroll
    for (int i = 0; i < 4; ++i) {                      // prologue: stage kk=0
        int c = c0 + i * 32;
        async_g2l_b128(wth + (size_t)c * FIN + ch, &s_w[0][c * 32 + ch]);
    }

    // prefetch first A fragment (K 0..15 / 16..31 by lane half)
    v16bf aNext = cat8(*(const v8bf*)(xrow), *(const v8bf*)(xrow + 8));

    v8f acc[8];
#pragma unroll
    for (int t = 0; t < 8; ++t)
#pragma unroll
        for (int r = 0; r < 8; ++r) acc[t][r] = 0.f;

    int cur = 0;
    for (int kk = 0; kk < FIN; kk += 32) {
        wait_async0();
        __syncthreads();

        if (kk + 32 < FIN) {     // overlap next Wt tile DMA with this step
#pragma unroll
            for (int i = 0; i < 4; ++i) {
                int c = c0 + i * 32;
                async_g2l_b128(wth + (size_t)c * FIN + (kk + 32) + ch,
                               &s_w[cur ^ 1][c * 32 + ch]);
            }
        }

        v16bf afrag = aNext;
        if (kk + 32 < FIN)       // prefetch next A fragment (registers)
            aNext = cat8(*(const v8bf*)(xrow + kk + 32),
                         *(const v8bf*)(xrow + kk + 40));

        // B fragments: col = t*16+lm, K = half*16 + 0..15 (contiguous 32B LDS)
        v16bf bfr[8];
#pragma unroll
        for (int t = 0; t < 8; ++t)
            bfr[t] = *(const v16bf*)(&s_w[cur][(t * 16 + lm) * 32 + half * 16]);
        sched_fence();
#pragma unroll
        for (int t = 0; t < 8; ++t)
            acc[t] = __builtin_amdgcn_wmma_f32_16x16x32_bf16(
                false, afrag, false, bfr[t], (short)0, acc[t], false, false);

        __syncthreads();
        cur ^= 1;
    }

    // f_src = h @ a[:FH], f_dst = h @ a[FH:], from f32 accumulators.
    // C layout: element (m = r + 8*half, n = lm) in acc[t][r], col = t*16+lm.
    float ps[8], pd[8];
#pragma unroll
    for (int r = 0; r < 8; ++r) { ps[r] = 0.f; pd[r] = 0.f; }
#pragma unroll
    for (int t = 0; t < 8; ++t) {
        float as = a[head * 2 * FH + t * 16 + lm];
        float ad = a[head * 2 * FH + FH + t * 16 + lm];
#pragma unroll
        for (int r = 0; r < 8; ++r) {
            ps[r] += acc[t][r] * as;
            pd[r] += acc[t][r] * ad;
        }
    }
#pragma unroll
    for (int r = 0; r < 8; ++r) {
        float s = ps[r], d = pd[r];
#pragma unroll
        for (int off = 1; off < 16; off <<= 1) {   // reduce within 16-lane half
            s += __shfl_xor(s, off, 32);
            d += __shfl_xor(d, off, 32);
        }
        if (lm == 0) {
            int row = row0 + r + 8 * half;
            fsrc[head * NN + row] = s;
            fdst[head * NN + row] = d;
        }
    }

    // store h transposed as bf16: hbT[head][col][row]  (K-contiguous for kernel B)
    __bf16* hb = hbT + (size_t)head * FH * NN;
#pragma unroll
    for (int t = 0; t < 8; ++t)
#pragma unroll
        for (int r = 0; r < 8; ++r) {
            int row = row0 + r + 8 * half;
            int col = t * 16 + lm;
            hb[(size_t)col * NN + row] = (__bf16)acc[t][r];
        }
}

// ------------- kernel B: fused attention  out = ELU(softmax(e) @ h) --------
// B tile (128 cols x 32 K = 8KB bf16) double-buffered in LDS via async loads.
__global__ __launch_bounds__(128)
void gat_attn_kernel(const __bf16* __restrict__ hbT, const float* __restrict__ fsrc,
                     const float* __restrict__ fdst, float* __restrict__ out) {
    __shared__ float s_d[NN];                        // 16 KB: f_dst for this head
    __shared__ float s_red[4];
    __shared__ __align__(64) __bf16 s_b[2][FH * 32]; // 2 x 8 KB ping-pong B tiles

    const int tid  = threadIdx.x;
    const int lane = tid & 31;
    const int wave = tid >> 5;
    const int head = blockIdx.y;
    const int row0 = blockIdx.x * 64 + wave * 16;
    const int half = lane >> 4;
    const int lm   = lane & 15;

    const __bf16* hb = hbT + (size_t)head * FH * NN;

    // --- prologue: async-stage first B tile (each thread copies 4 x 16B) ---
    const int c0 = tid >> 2, ch = (tid & 3) * 8;
#pragma unroll
    for (int i = 0; i < 4; ++i) {
        int c = c0 + i * 32;
        async_g2l_b128(hb + (size_t)c * NN + ch, &s_b[0][c * 32 + ch]);
    }

    // stage f_dst in LDS + compute max_j f_dst (closed-form row max)
    float mymax = -1e30f;
    for (int j = tid; j < NN; j += 128) {
        float v = fdst[head * NN + j];
        s_d[j] = v;
        mymax = fmaxf(mymax, v);
    }
#pragma unroll
    for (int off = 16; off >= 1; off >>= 1)
        mymax = fmaxf(mymax, __shfl_xor(mymax, off, 32));
    if (lane == 0) s_red[wave] = mymax;
    __syncthreads();
    const float maxd = fmaxf(fmaxf(s_red[0], s_red[1]), fmaxf(s_red[2], s_red[3]));

    const float si = fsrc[head * NN + row0 + lm];      // A-row for this lane
    const float t0 = si + maxd;
    const float mi = t0 > 0.f ? t0 : LRELU * t0;       // row max (LR monotone)

    v8f acc[8];
#pragma unroll
    for (int t = 0; t < 8; ++t)
#pragma unroll
        for (int r = 0; r < 8; ++r) acc[t][r] = 0.f;
    float lsum = 0.f;

    int cur = 0;
    for (int jj = 0; jj < NN; jj += 32) {
        wait_async0();          // this wave's copies into s_b[cur] done
        __syncthreads();        // all waves' copies done; prev reads finished

        if (jj + 32 < NN) {     // overlap next tile's DMA with this step's math
#pragma unroll
            for (int i = 0; i < 4; ++i) {
                int c = c0 + i * 32;
                async_g2l_b128(hb + (size_t)c * NN + (jj + 32) + ch,
                               &s_b[cur ^ 1][c * 32 + ch]);
            }
        }

        // A fragment: p_ij = exp(LeakyReLU(s_i + d_j) - m_i), 16 per lane
        const float* dp = s_d + jj + half * 16;
        float pv[16];
#pragma unroll
        for (int q = 0; q < 8; ++q) {
            float e0 = si + dp[q];        // K = jj + half*16 + q
            float e1 = si + dp[q + 8];    // K = jj + half*16 + 8 + q
            e0 = e0 > 0.f ? e0 : LRELU * e0;
            e1 = e1 > 0.f ? e1 : LRELU * e1;
            pv[q]     = __expf(e0 - mi);
            pv[q + 8] = __expf(e1 - mi);
        }
        v16bf afrag;
#pragma unroll
        for (int q = 0; q < 16; ++q) { afrag[q] = (__bf16)pv[q]; lsum += pv[q]; }

        // B fragments from LDS (contiguous 32B per lane), then 8 clean WMMAs
        v16bf bfr[8];
#pragma unroll
        for (int t = 0; t < 8; ++t)
            bfr[t] = *(const v16bf*)(&s_b[cur][(t * 16 + lm) * 32 + half * 16]);
        sched_fence();
#pragma unroll
        for (int t = 0; t < 8; ++t)
            acc[t] = __builtin_amdgcn_wmma_f32_16x16x32_bf16(
                false, afrag, false, bfr[t], (short)0, acc[t], false, false);

        __syncthreads();        // everyone done reading s_b[cur] before refill
        cur ^= 1;
    }

    lsum += __shfl_xor(lsum, 16, 32);   // lane L now has row total for row L&15

#pragma unroll
    for (int r = 0; r < 8; ++r) {
        float inv = 1.f / __shfl(lsum, r + 8 * half, 32);  // row = r + 8*half
        int row = row0 + r + 8 * half;
        float* orow = out + (size_t)row * (NH * FH) + head * FH + lm;
#pragma unroll
        for (int t = 0; t < 8; ++t) {
            float o = acc[t][r] * inv;
            o = o > 0.f ? o : (__expf(o) - 1.f);           // ELU (alpha=1)
            orow[t * 16] = o;                               // coalesced per 16 lanes
        }
    }
}

// ------------------------------ launcher -----------------------------------
extern "C" void kernel_launch(void* const* d_in, const int* in_sizes, int n_in,
                              void* d_out, int out_size, void* d_ws, size_t ws_size,
                              hipStream_t stream) {
    (void)in_sizes; (void)n_in; (void)out_size; (void)ws_size;
    const float* x = (const float*)d_in[0];
    const float* W = (const float*)d_in[1];
    const float* a = (const float*)d_in[2];
    float* out = (float*)d_out;

    // workspace layout (~8.7 MB total)
    __bf16* xb  = (__bf16*)d_ws;                        // NN*FIN      bf16 (4 MB)
    __bf16* wt  = xb + (size_t)NN * FIN;                // NH*FH*FIN   bf16 (0.5 MB)
    __bf16* hbT = wt + (size_t)NH * FH * FIN;           // NH*FH*NN    bf16 (4 MB)
    float*  fsrc = (float*)(hbT + (size_t)NH * FH * NN);// NH*NN f32
    float*  fdst = fsrc + NH * NN;                      // NH*NN f32

    prep_x_kernel<<<512, 256, 0, stream>>>(x, xb);
    prep_w_kernel<<<256, 256, 0, stream>>>(W, wt);

    dim3 grid(NN / 64, NH), block(128);
    gat_h_kernel<<<grid, block, 0, stream>>>(xb, wt, a, hbT, fsrc, fdst);
    gat_attn_kernel<<<grid, block, 0, stream>>>(hbT, fsrc, fdst, out);
}
